// TransformerEncoder_86818468922009
// MI455X (gfx1250) — matrix operand order repaired and checked by reading, not compiled, not run on hardware
//
#include <hip/hip_runtime.h>
#include <hip/hip_bf16.h>

typedef __attribute__((ext_vector_type(16))) __bf16 v16bf;
typedef __attribute__((ext_vector_type(8)))  float  v8f;

#define D_MODEL    512
#define NUM_HEADS  8
#define NUM_LAYERS 4
#define SEQ_LEN    2048
#define BATCH      2
#define LN_EPS     1e-5f

// Fragment-pair storage: 32 lanes x (16 bf16 hi + 16 bf16 lo) = 1024 bf16 = 2KB.
// Per-lane offset lane*32 elems; hi at +0..15, lo at +16..31.
#define FRAG_ELEMS 1024

// ---------------------------------------------------------------------------
// bf16 split: x = hi + lo (bf16x3 scheme gives ~fp32 fidelity with fp32 accum)
// ---------------------------------------------------------------------------
__device__ inline void split16(const float* v, v16bf& hi, v16bf& lo) {
#pragma unroll
  for (int j = 0; j < 16; ++j) {
    float x  = v[j];
    __bf16 h = (__bf16)x;
    hi[j] = h;
    lo[j] = (__bf16)(x - (float)h);
  }
}

__device__ inline v8f wmma3(v16bf ahi, v16bf alo, v16bf bhi, v16bf blo, v8f c) {
  c = __builtin_amdgcn_wmma_f32_16x16x32_bf16(false, ahi, false, bhi, (short)0, c, false, false);
  c = __builtin_amdgcn_wmma_f32_16x16x32_bf16(false, ahi, false, blo, (short)0, c, false, false);
  c = __builtin_amdgcn_wmma_f32_16x16x32_bf16(false, alo, false, bhi, (short)0, c, false, false);
  return c;
}

// A fragment (16x32, this file's contiguous k order): lane m=lane&15, half=lane>>4 holds
// K = half*16 + 0..15 (elems 0..15).
__device__ inline void load_frag_a(const float* p /* row m, col k0+half*16 */, float* v) {
  *(float4*)(v)      = *(const float4*)(p);
  *(float4*)(v + 4)  = *(const float4*)(p + 4);
  *(float4*)(v + 8)  = *(const float4*)(p + 8);
  *(float4*)(v + 12) = *(const float4*)(p + 12);
}

__device__ inline void store_frag(__bf16* base, int lane, v16bf hi, v16bf lo) {
  __bf16* o = base + lane * 32;
  *(v16bf*)(o)      = hi;
  *(v16bf*)(o + 16) = lo;
}

// ---------------------------------------------------------------------------
// Pack fp32 row-major A[M,K] -> A-fragment pairs, frag id = mt*ktiles + kt
// ---------------------------------------------------------------------------
__global__ __launch_bounds__(256) void pack_a_kernel(
    const float* __restrict__ A, __bf16* __restrict__ P, int M, int K) {
  const int wave = blockIdx.x * 8 + (threadIdx.x >> 5);
  const int ktiles = K >> 5;
  if (wave >= (M >> 4) * ktiles) return;
  const int mt = wave / ktiles, kt = wave % ktiles;
  const int lane = threadIdx.x & 31, am = lane & 15, ah = lane >> 4;
  float v[16];
  load_frag_a(A + (size_t)(mt * 16 + am) * K + kt * 32 + ah * 16, v);
  v16bf hi, lo; split16(v, hi, lo);
  store_frag(P + (size_t)wave * FRAG_ELEMS, lane, hi, lo);
}

// ---------------------------------------------------------------------------
// Pack fp32 row-major B[K,N] -> B-fragment pairs, frag id = kt*(N/16) + nt.
// B layout: lane n=lane&15, half=lane>>4 holds K = half*16 + 0..15, col n.
// ---------------------------------------------------------------------------
__global__ __launch_bounds__(256) void pack_b_kernel(
    const float* __restrict__ B, __bf16* __restrict__ P, int K, int N) {
  const int wave = blockIdx.x * 8 + (threadIdx.x >> 5);
  const int n16 = N >> 4;
  if (wave >= (K >> 5) * n16) return;
  const int kt = wave / n16, nt = wave % n16;
  const int lane = threadIdx.x & 31, am = lane & 15, ah = lane >> 4;
  const float* p = B + (size_t)(kt * 32 + ah * 16) * N + nt * 16 + am;
  float v[16];
#pragma unroll
  for (int j = 0; j < 16; ++j) v[j] = p[(size_t)j * N];
  v16bf hi, lo; split16(v, hi, lo);
  store_frag(P + (size_t)wave * FRAG_ELEMS, lane, hi, lo);
}

// ---------------------------------------------------------------------------
// Pack Q (A-layout) / K (B=K^T layout) head slices of qkv.
// Frag id = ((b*H + h)*Ltiles + tile)*2 + s   (s = k-step over dh=64)
// ---------------------------------------------------------------------------
__global__ __launch_bounds__(256) void pack_q_kernel(
    const float* __restrict__ qkv, __bf16* __restrict__ P) {
  const int wave = blockIdx.x * 8 + (threadIdx.x >> 5);
  const int total = BATCH * NUM_HEADS * (SEQ_LEN / 16) * 2;
  if (wave >= total) return;
  const int s = wave & 1;
  int t = wave >> 1;
  const int lt = t & 127; t >>= 7;
  const int h = t & 7, b = t >> 3;
  const int lane = threadIdx.x & 31, am = lane & 15, ah = lane >> 4;
  const float* p = qkv + (size_t)(b * SEQ_LEN + lt * 16 + am) * (3 * D_MODEL)
                 + h * 64 + s * 32 + ah * 16;
  float v[16];
  load_frag_a(p, v);
  v16bf hi, lo; split16(v, hi, lo);
  store_frag(P + (size_t)wave * FRAG_ELEMS, lane, hi, lo);
}

__global__ __launch_bounds__(256) void pack_k_kernel(
    const float* __restrict__ qkv, __bf16* __restrict__ P) {
  const int wave = blockIdx.x * 8 + (threadIdx.x >> 5);
  const int total = BATCH * NUM_HEADS * (SEQ_LEN / 16) * 2;
  if (wave >= total) return;
  const int s = wave & 1;
  int t = wave >> 1;
  const int kt = t & 127; t >>= 7;
  const int h = t & 7, b = t >> 3;
  const int lane = threadIdx.x & 31, am = lane & 15, ah = lane >> 4;
  // B_gemm[j, n] = Kmat[key n, j]: lane am = key row, contiguous j run.
  const float* p = qkv + (size_t)(b * SEQ_LEN + kt * 16 + am) * (3 * D_MODEL)
                 + D_MODEL + h * 64 + s * 32 + ah * 16;
  float v[16];
  *(float4*)(v)      = *(const float4*)(p);
  *(float4*)(v + 4)  = *(const float4*)(p + 4);
  *(float4*)(v + 8)  = *(const float4*)(p + 8);
  *(float4*)(v + 12) = *(const float4*)(p + 12);
  v16bf hi, lo; split16(v, hi, lo);
  store_frag(P + (size_t)wave * FRAG_ELEMS, lane, hi, lo);
}

// ---------------------------------------------------------------------------
// GEMM on packed fragments: C[M,N] = A @ B. One wave owns a 64x32 macro-tile
// (4x2 WMMA tiles): per K-step 24 b128 loads : 24 WMMAs, zero conversion VALU.
// EPI: 0 none, 1 +bias, 2 +bias then PReLU(alpha).
// ---------------------------------------------------------------------------
template <int EPI>
__global__ __launch_bounds__(128) void gemm_packed(
    const __bf16* __restrict__ PA, const __bf16* __restrict__ PB,
    float* __restrict__ C, const float* __restrict__ bias,
    const float* __restrict__ alpha_ptr, int M, int N, int K) {
  const int lane = threadIdx.x & 31;
  const int wave = blockIdx.x * (blockDim.x >> 5) + (threadIdx.x >> 5);
  const int nt32 = N >> 5;
  if (wave >= (M >> 6) * nt32) return;
  const int m0 = (wave / nt32) << 6;
  const int n0 = (wave % nt32) << 5;
  const int ktiles = K >> 5;
  const int n16 = N >> 4;
  const int am = lane & 15, ah = lane >> 4;

  const __bf16* pa = PA + (size_t)(m0 >> 4) * ktiles * FRAG_ELEMS + lane * 32;
  const __bf16* pb = PB + (size_t)(n0 >> 4) * FRAG_ELEMS + lane * 32;

  v8f acc[4][2] = {};
  for (int kt = 0; kt < ktiles; ++kt) {
    v16bf Ahi[4], Alo[4], Bhi[2], Blo[2];
#pragma unroll
    for (int mi = 0; mi < 4; ++mi) {
      const __bf16* p = pa + ((size_t)mi * ktiles + kt) * FRAG_ELEMS;
      Ahi[mi] = *(const v16bf*)p;
      Alo[mi] = *(const v16bf*)(p + 16);
    }
#pragma unroll
    for (int ni = 0; ni < 2; ++ni) {
      const __bf16* p = pb + ((size_t)kt * n16 + ni) * FRAG_ELEMS;
      Bhi[ni] = *(const v16bf*)p;
      Blo[ni] = *(const v16bf*)(p + 16);
    }
#pragma unroll
    for (int mi = 0; mi < 4; ++mi)
#pragma unroll
      for (int ni = 0; ni < 2; ++ni)
        acc[mi][ni] = wmma3(Ahi[mi], Alo[mi], Bhi[ni], Blo[ni], acc[mi][ni]);
  }

  const float alpha = (EPI == 2) ? alpha_ptr[0] : 0.f;
#pragma unroll
  for (int mi = 0; mi < 4; ++mi)
#pragma unroll
    for (int ni = 0; ni < 2; ++ni) {
      const int col = n0 + ni * 16 + am;
      const float bv = (EPI >= 1) ? bias[col] : 0.f;
#pragma unroll
      for (int r = 0; r < 8; ++r) {
        // C layout: vgpr r -> row = 8*half + r, col = lane&15
        const int row = m0 + mi * 16 + ah * 8 + r;
        float v = acc[mi][ni][r] + bv;
        if (EPI == 2) v = (v >= 0.f) ? v : alpha * v;
        C[(size_t)row * N + col] = v;
      }
    }
}

// ---------------------------------------------------------------------------
// Attention: reference keeps only the DIAGONAL of softmax(QK^T*scale+mask)
// and scales V elementwise. One wave per (b,h,16-query tile). The diagonal
// logit is computed ONCE in fp32 outside the streaming loop (one 64-dot per
// row), so the hot loop is just frag loads + 6 WMMAs + online max/sum.
// ---------------------------------------------------------------------------
__global__ __launch_bounds__(32) void attn_diag_kernel(
    const __bf16* __restrict__ PQ, const __bf16* __restrict__ PK,
    const float* __restrict__ qkv, const float* __restrict__ mask,
    float* __restrict__ out) {
  const int L = SEQ_LEN, d = D_MODEL, dh = D_MODEL / NUM_HEADS;  // 64
  const int lane = threadIdx.x & 31;
  int bid = blockIdx.x;
  const int ltiles = L / 16;
  const int lt = bid % ltiles; bid /= ltiles;
  const int h = bid % NUM_HEADS;
  const int b = bid / NUM_HEADS;
  const int l0 = lt * 16;
  const int am = lane & 15, ah = lane >> 4;
  const float scale = 0.044194173824159216f;  // 1/sqrt(d_model)
  const int bh = b * NUM_HEADS + h;

  // fp32 diagonal logit for row = l0 + am (each 16-lane group covers 16 rows)
  float dg;
  {
    const int row = l0 + am;
    const float* qp = qkv + (size_t)(b * L + row) * (3 * d) + h * dh;
    const float* kp = qp + d;
    float acc = 0.f;
#pragma unroll
    for (int j = 0; j < 64; j += 4) {
      float4 qv = *(const float4*)(qp + j);
      float4 kv = *(const float4*)(kp + j);
      acc = fmaf(qv.x, kv.x, acc);
      acc = fmaf(qv.y, kv.y, acc);
      acc = fmaf(qv.z, kv.z, acc);
      acc = fmaf(qv.w, kv.w, acc);
    }
    dg = acc * scale + mask[(size_t)row * L + row];
  }

  v16bf qhi[2], qlo[2];
  {
    const __bf16* p = PQ + ((size_t)(bh * ltiles + lt) * 2) * FRAG_ELEMS + lane * 32;
    qhi[0] = *(const v16bf*)p;                qlo[0] = *(const v16bf*)(p + 16);
    qhi[1] = *(const v16bf*)(p + FRAG_ELEMS); qlo[1] = *(const v16bf*)(p + FRAG_ELEMS + 16);
  }

  float mrun[8], srun[8];
#pragma unroll
  for (int r = 0; r < 8; ++r) { mrun[r] = -3.0e38f; srun[r] = 0.f; }

  const __bf16* pk = PK + ((size_t)bh * ltiles * 2) * FRAG_ELEMS + lane * 32;
  const float* mrow = mask + (size_t)(l0 + ah * 8) * L + am;
  for (int kt = 0; kt < ltiles; ++kt) {
    const __bf16* p = pk + (size_t)kt * 2 * FRAG_ELEMS;
    v16bf k0h = *(const v16bf*)p,                k0l = *(const v16bf*)(p + 16);
    v16bf k1h = *(const v16bf*)(p + FRAG_ELEMS), k1l = *(const v16bf*)(p + FRAG_ELEMS + 16);
    v8f acc = {};
    acc = wmma3(qhi[0], qlo[0], k0h, k0l, acc);
    acc = wmma3(qhi[1], qlo[1], k1h, k1l, acc);

    const float* mp = mrow + kt * 16;  // + r*L per row below
#pragma unroll
    for (int r = 0; r < 8; ++r) {
      float v = acc[r] * scale + mp[(size_t)r * L];
      float mn = fmaxf(mrun[r], v);
      srun[r]  = srun[r] * __expf(mrun[r] - mn) + __expf(v - mn);
      mrun[r]  = mn;
    }
  }

#pragma unroll
  for (int r = 0; r < 8; ++r) {
#pragma unroll
    for (int off = 1; off <= 8; off <<= 1) {  // reduce within each 16-lane group
      float mo = __shfl_xor(mrun[r], off, 32);
      float so = __shfl_xor(srun[r], off, 32);
      float mn = fmaxf(mrun[r], mo);
      srun[r]  = srun[r] * __expf(mrun[r] - mn) + so * __expf(mo - mn);
      mrun[r]  = mn;
    }
    // diagonal logit for row ah*8+r lives in group lane (ah*8+r)
    const float dgr = __shfl(dg, (lane & 16) | (ah * 8 + r), 32);
    const float a = (srun[r] > 0.f) ? __expf(dgr - mrun[r]) / srun[r] : 0.f;
    const int row = l0 + ah * 8 + r;
    const float* vp = qkv + (size_t)(b * L + row) * (3 * d) + 2 * d + h * dh + am * 4;
    float* op = out + (size_t)(b * L + row) * d + h * dh + am * 4;
    float4 vv = *(const float4*)vp;
    vv.x *= a; vv.y *= a; vv.z *= a; vv.w *= a;
    *(float4*)op = vv;
  }
}

// ---------------------------------------------------------------------------
// LayerNorm (optionally + residual): one wave per row of 512.
// ---------------------------------------------------------------------------
__global__ __launch_bounds__(256) void ln_kernel(
    const float* __restrict__ in, const float* __restrict__ res,
    const float* __restrict__ g, const float* __restrict__ bta,
    float* __restrict__ out, int rows) {
  const int d = D_MODEL;
  const int lane = threadIdx.x & 31;
  const int row  = blockIdx.x * (blockDim.x >> 5) + (threadIdx.x >> 5);
  if (row >= rows) return;
  const float* ip = in + (size_t)row * d;
  float v[16];
  float s = 0.f;
#pragma unroll
  for (int i = 0; i < 16; ++i) {
    float x = ip[lane + i * 32];
    if (res) x += res[(size_t)row * d + lane + i * 32];
    v[i] = x; s += x;
  }
#pragma unroll
  for (int off = 1; off <= 16; off <<= 1) s += __shfl_xor(s, off, 32);
  const float mu = s * (1.f / 512.f);
  float s2 = 0.f;
#pragma unroll
  for (int i = 0; i < 16; ++i) { float t = v[i] - mu; s2 += t * t; }
#pragma unroll
  for (int off = 1; off <= 16; off <<= 1) s2 += __shfl_xor(s2, off, 32);
  const float inv = rsqrtf(s2 * (1.f / 512.f) + LN_EPS);
#pragma unroll
  for (int i = 0; i < 16; ++i)
    out[(size_t)row * d + lane + i * 32] =
        (v[i] - mu) * inv * g[lane + i * 32] + bta[lane + i * 32];
}

// ---------------------------------------------------------------------------
// Launcher
// ---------------------------------------------------------------------------
extern "C" void kernel_launch(void* const* d_in, const int* in_sizes, int n_in,
                              void* d_out, int out_size, void* d_ws, size_t ws_size,
                              hipStream_t stream) {
  (void)in_sizes; (void)n_in; (void)out_size; (void)ws_size;
  const float* x     = (const float*)d_in[0];
  const float* mask  = (const float*)d_in[1];
  const float* Wqkv  = (const float*)d_in[2];
  const float* Wout  = (const float*)d_in[3];
  const float* ln_g  = (const float*)d_in[4];
  const float* ln_b  = (const float*)d_in[5];
  const float* W1    = (const float*)d_in[6];
  const float* b1    = (const float*)d_in[7];
  const float* alpha = (const float*)d_in[8];
  const float* W2    = (const float*)d_in[9];
  const float* b2    = (const float*)d_in[10];
  const float* lnf_g = (const float*)d_in[11];
  const float* lnf_b = (const float*)d_in[12];
  float* outp = (float*)d_out;

  const int R = BATCH * SEQ_LEN;  // 4096 rows
  const int d = D_MODEL;

  // fp32 scratch
  float* ws      = (float*)d_ws;
  float* x_cur   = ws; ws += (size_t)R * d;
  float* qkv     = ws; ws += (size_t)R * 3 * d;
  float* bufA    = ws; ws += (size_t)R * d;
  float* bufB    = ws; ws += (size_t)R * d;
  float* out_att = ws; ws += (size_t)R * d;
  float* hbuf    = ws; ws += (size_t)R * 4 * d;
  // bf16 fragment scratch (2 bf16 per source element)
  __bf16* bw     = (__bf16*)ws;
  __bf16* packA  = bw; bw += (size_t)R * 4 * d * 2;           // up to 4096x2048
  __bf16* packB  = bw; bw += (size_t)4 * d * d * 2;           // up to 2048x512 / 512x2048
  __bf16* packQ  = bw; bw += (size_t)R * (d / NUM_HEADS) * 2; // B*L*dh
  __bf16* packK  = bw; bw += (size_t)R * (d / NUM_HEADS) * 2;

  hipMemcpyAsync(x_cur, x, (size_t)R * d * sizeof(float),
                 hipMemcpyDeviceToDevice, stream);

  auto pa_blocks = [](int M, int K) { int w = (M / 16) * (K / 32); return (w + 7) / 8; };
  auto pb_blocks = [](int K, int N) { int w = (K / 32) * (N / 16); return (w + 7) / 8; };
  auto gm_blocks = [](int M, int N) { int w = (M / 64) * (N / 32); return (w + 3) / 4; };
  const int qk_blocks = (BATCH * NUM_HEADS * (SEQ_LEN / 16) * 2 + 7) / 8;

  for (int l = 0; l < NUM_LAYERS; ++l) {
    const float* wqkv = Wqkv + (size_t)l * d * 3 * d;
    const float* wout = Wout + (size_t)l * d * d;
    const float* g    = ln_g + (size_t)l * d;
    const float* bln  = ln_b + (size_t)l * d;
    const float* w1   = W1 + (size_t)l * d * 4 * d;
    const float* bb1  = b1 + (size_t)l * 4 * d;
    const float* al   = alpha + l;
    const float* w2   = W2 + (size_t)l * 4 * d * d;
    const float* bb2  = b2 + (size_t)l * d;

    // qkv = x @ Wqkv
    pack_a_kernel<<<pa_blocks(R, d), 256, 0, stream>>>(x_cur, packA, R, d);
    pack_b_kernel<<<pb_blocks(d, 3 * d), 256, 0, stream>>>(wqkv, packB, d, 3 * d);
    gemm_packed<0><<<gm_blocks(R, 3 * d), 128, 0, stream>>>(
        packA, packB, qkv, nullptr, nullptr, R, 3 * d, d);

    // diagonal-softmax attention
    pack_q_kernel<<<qk_blocks, 256, 0, stream>>>(qkv, packQ);
    pack_k_kernel<<<qk_blocks, 256, 0, stream>>>(qkv, packK);
    attn_diag_kernel<<<BATCH * NUM_HEADS * (SEQ_LEN / 16), 32, 0, stream>>>(
        packQ, packK, qkv, mask, bufA);

    // out = attn @ Wout ; out_att = LN(out + x)
    pack_a_kernel<<<pa_blocks(R, d), 256, 0, stream>>>(bufA, packA, R, d);
    pack_b_kernel<<<pb_blocks(d, d), 256, 0, stream>>>(wout, packB, d, d);
    gemm_packed<0><<<gm_blocks(R, d), 128, 0, stream>>>(
        packA, packB, bufB, nullptr, nullptr, R, d, d);
    ln_kernel<<<(R + 7) / 8, 256, 0, stream>>>(bufB, x_cur, g, bln, out_att, R);

    // h = PReLU(out_att @ W1 + b1)
    pack_a_kernel<<<pa_blocks(R, d), 256, 0, stream>>>(out_att, packA, R, d);
    pack_b_kernel<<<pb_blocks(d, 4 * d), 256, 0, stream>>>(w1, packB, d, 4 * d);
    gemm_packed<2><<<gm_blocks(R, 4 * d), 128, 0, stream>>>(
        packA, packB, hbuf, bb1, al, R, 4 * d, d);

    // x = LN(h @ W2 + b2 + out_att)
    pack_a_kernel<<<pa_blocks(R, 4 * d), 256, 0, stream>>>(hbuf, packA, R, 4 * d);
    pack_b_kernel<<<pb_blocks(4 * d, d), 256, 0, stream>>>(w2, packB, 4 * d, d);
    gemm_packed<1><<<gm_blocks(R, d), 128, 0, stream>>>(
        packA, packB, bufB, bb2, nullptr, R, d, 4 * d);
    ln_kernel<<<(R + 7) / 8, 256, 0, stream>>>(bufB, out_att, g, bln, x_cur, R);
  }
  ln_kernel<<<(R + 7) / 8, 256, 0, stream>>>(x_cur, nullptr, lnf_g, lnf_b, outp, R);
}
